// AdaptiveConv1d_38448547234212
// MI455X (gfx1250) — matrix operand order repaired and checked by reading, not compiled, not run on hardware
//
#include <hip/hip_runtime.h>
#include <stdint.h>

// ---------------------------------------------------------------------------
// AdaptiveConv1d as one GEMM: out = [W0|W1|W2] @ [xP;xC;xF] + b0
// fp32 via bf16 hi/lo split (3 WMMA products) on v_wmma_f32_16x16x32_bf16.
// A (weights, pre-split to bf16 in scratch) streamed to LDS by the Tensor
// Data Mover (tensor_load_to_lds, TENSORcnt), double-buffered against WMMA.
// B (gathered x) pipelined: global f32 loads for chunk c+1 issued before the
// chunk-c WMMA block; converted to bf16 hi/lo at fragment build (co-executes
// with WMMA).
// ---------------------------------------------------------------------------

#define CIN   512
#define COUT  512
#define TLEN  8192
#define BATCH 8

#define BM 128   // Cout rows per workgroup
#define BN 128   // t columns per workgroup
#define BK 32    // K chunk (bf16 WMMA K)
#define NCHUNK (3 * CIN / BK)   // 48

typedef __bf16 bf16x16 __attribute__((ext_vector_type(16)));
typedef float  f32x8   __attribute__((ext_vector_type(8)));
typedef unsigned int u32x4 __attribute__((ext_vector_type(4)));
typedef int          i32x4 __attribute__((ext_vector_type(4)));
typedef int          i32x8 __attribute__((ext_vector_type(8)));

__device__ __forceinline__ unsigned short f2bf_rne(float f) {
  unsigned int u = __float_as_uint(f);
  u += 0x7FFFu + ((u >> 16) & 1u);           // round-to-nearest-even
  return (unsigned short)(u >> 16);
}
__device__ __forceinline__ float bf2f(unsigned short h) {
  return __uint_as_float(((unsigned int)h) << 16);
}

// ---------------------------------------------------------------------------
// Prep: split W0/W1/W2 (fp32) into bf16 hi + bf16 lo residual in scratch.
// ---------------------------------------------------------------------------
__global__ void prep_weights_kernel(const float* __restrict__ W0,
                                    const float* __restrict__ W1,
                                    const float* __restrict__ W2,
                                    unsigned short* __restrict__ whi,
                                    unsigned short* __restrict__ wlo) {
  int i = blockIdx.x * blockDim.x + threadIdx.x;
  const int N1 = COUT * CIN;
  if (i >= 3 * N1) return;
  const float* W = (i < N1) ? W0 : ((i < 2 * N1) ? W1 : W2);
  float w = W[i % N1];
  unsigned short h = f2bf_rne(w);
  whi[i] = h;
  wlo[i] = f2bf_rne(w - bf2f(h));
}

// ---------------------------------------------------------------------------
// TDM: DMA one 128(row) x 32(col) bf16 tile (row pitch CIN) global -> LDS.
// D# per CDNA5 ISA ch.8: group0 = {count/flags, lds_addr, global_addr, type},
// group1 = {data_size, tensor dims/strides, tile dims}. 2D: groups 2/3 zero.
// This toolchain exposes the 6-arg builtin: (g0, g1, g2, g3, g_extra, cpol).
// ---------------------------------------------------------------------------
__device__ __forceinline__ void tdm_load_tile_bf16(const unsigned short* gsrc,
                                                   unsigned lds_byte_off) {
  unsigned long long ga = (unsigned long long)(uintptr_t)gsrc;
  u32x4 g0;
  g0[0] = 1u;                                        // count=1, no gather
  g0[1] = lds_byte_off;                              // lds_addr
  g0[2] = (unsigned)(ga & 0xFFFFFFFFu);              // global_addr[31:0]
  g0[3] = (unsigned)((ga >> 32) & 0x01FFFFFFu)       // global_addr[56:32]
        | (2u << 30);                                // type = 2 (image)
  i32x8 g1;
  g1[0] = 1 << 16;                                   // data_size = 1 (2B)
  g1[1] = (CIN & 0xFFFF) << 16;                      // tensor_dim0[15:0]
  g1[2] = (COUT & 0xFFFF) << 16;                     // dim0[31:16]=0 | dim1[15:0]
  g1[3] = BK << 16;                                  // dim1[31:16]=0 | tile_dim0
  g1[4] = BM;                                        // tile_dim1 | tile_dim2=0
  g1[5] = CIN;                                       // tensor_dim0_stride[31:0]
  g1[6] = 0;                                         // stride0 hi | stride1 lo
  g1[7] = 0;
  i32x4 z4 = (i32x4)0;
  i32x8 z8 = (i32x8)0;
  __builtin_amdgcn_tensor_load_to_lds(g0, g1, z4, z4, z8, 0);
}

// ---------------------------------------------------------------------------
// Main GEMM kernel. Grid: (BATCH * TLEN/BN, COUT/BM), 256 threads (8 wave32).
// ---------------------------------------------------------------------------
__global__ __launch_bounds__(256)
void adaptive_conv_wmma_kernel(const float* __restrict__ x,
                               const float* __restrict__ dfac,
                               const unsigned short* __restrict__ whi,
                               const unsigned short* __restrict__ wlo,
                               const float* __restrict__ bias,
                               float* __restrict__ out) {
  // double-buffered LDS tiles
  __shared__ alignas(16) unsigned short Ahi[2][BM * BK];   // bf16 [m][k]
  __shared__ alignas(16) unsigned short Alo[2][BM * BK];   // bf16 [m][k]
  __shared__ alignas(16) float          Bf32[2][BN * BK];  // f32  [n][k]
  __shared__ int   colP[BN];
  __shared__ int   colF[BN];
  __shared__ float biasS[BM];

  const int tid  = threadIdx.x;
  const int lane = tid & 31;          // wave32
  const int wave = tid >> 5;          // 0..7
  const int wm   = wave >> 2;         // 0..1 (M)
  const int wn   = wave & 3;          // 0..3 (N)
  const int half = lane >> 4;         // 0/1 lane group
  const int lr   = lane & 15;

  const int nblk = blockIdx.x;
  const int b    = nblk / (TLEN / BN);
  const int t0   = (nblk % (TLEN / BN)) * BN;
  const int m0   = blockIdx.y * BM;

  // ---- gather-index prologue (reference semantics, DILATION = 1) ----
  if (tid < BN) {
    int t = t0 + tid;
    float dv  = dfac[(size_t)b * TLEN + t];
    int   dil = (int)(dv * 1.0f);
    if (dil < 1) dil = 1;
    int ip = t - dil; if (ip < 0) ip = -ip;
    ip &= (TLEN - 1);                               // abs(t-dil) % T
    int f0 = t + dil;
    int fi = (f0 >= TLEN) ? (TLEN - 1 - (f0 & (TLEN - 1))) : f0;
    colP[tid] = ip;
    colF[tid] = fi;
  }
  if (tid < BM) biasS[tid] = bias[m0 + tid];
  __syncthreads();                                   // colP/colF visible

  f32x8 acc[4][2];
  #pragma unroll
  for (int i = 0; i < 4; ++i)
    #pragma unroll
    for (int j = 0; j < 2; ++j) acc[i][j] = (f32x8)(0.0f);

  const float* xb = x + (size_t)b * CIN * TLEN;
  const unsigned short* wbh = whi + (size_t)m0 * CIN;  // + part*COUT*CIN later
  const unsigned short* wbl = wlo + (size_t)m0 * CIN;

  const int b_col = tid & 127;        // 0..127 (t column)
  const int b_q   = tid >> 7;         // 0..1   (k quarter interleave)

  // ---- prologue stage of chunk 0 ----
  if (wave == 0) {
    tdm_load_tile_bf16(wbh, (unsigned)(uintptr_t)&Ahi[0][0]);
    tdm_load_tile_bf16(wbl, (unsigned)(uintptr_t)&Alo[0][0]);
  }
  {
    int tc = colP[b_col];                            // chunk 0: part 0 (xP)
    const float* src = xb + tc;                      // kc = 0
    #pragma unroll
    for (int i = 0; i < 4; ++i) {
      int k = b_q * 4 + i * 8;
      float4 v;
      v.x = src[(size_t)(k + 0) * TLEN];
      v.y = src[(size_t)(k + 1) * TLEN];
      v.z = src[(size_t)(k + 2) * TLEN];
      v.w = src[(size_t)(k + 3) * TLEN];
      *(float4*)(&Bf32[0][b_col * BK + k]) = v;
    }
  }
  if (wave == 0) __builtin_amdgcn_s_wait_tensorcnt(0);
  __syncthreads();

  union FragA { bf16x16 v; uint4 u[2]; };
  union FragB { bf16x16 v; unsigned short s[16]; };
  union FragF { float f[16]; uint4 u[4]; };

  for (int c = 0; c < NCHUNK; ++c) {
    const int cur = c & 1;
    const int nxt = cur ^ 1;
    const bool has_next = (c + 1 < NCHUNK);

    // ---- issue next A tile DMA (TDM) ----
    if (wave == 0 && has_next) {
      int cn = c + 1;
      int partn = cn >> 4;
      int kcn   = (cn & 15) * BK;
      size_t off = (size_t)partn * COUT * CIN + kcn;
      tdm_load_tile_bf16(wbh + off, (unsigned)(uintptr_t)&Ahi[nxt][0]);
      tdm_load_tile_bf16(wbl + off, (unsigned)(uintptr_t)&Alo[nxt][0]);
    }

    // ---- issue next B global loads (land during WMMA below) ----
    float breg[16];
    if (has_next) {
      int cn = c + 1;
      int partn = cn >> 4;
      int kcn   = (cn & 15) * BK;
      int tc;
      if (partn == 0)      tc = colP[b_col];
      else if (partn == 1) tc = t0 + b_col;
      else                 tc = colF[b_col];
      const float* src = xb + (size_t)kcn * TLEN + tc;
      #pragma unroll
      for (int i = 0; i < 4; ++i) {
        int k = b_q * 4 + i * 8;
        breg[i * 4 + 0] = src[(size_t)(k + 0) * TLEN];
        breg[i * 4 + 1] = src[(size_t)(k + 1) * TLEN];
        breg[i * 4 + 2] = src[(size_t)(k + 2) * TLEN];
        breg[i * 4 + 3] = src[(size_t)(k + 3) * TLEN];
      }
    }

    // ---- A fragments: direct bf16 ds_load_b128 per ISA 16x32 layout ----
    FragA ah[4], al[4];
    #pragma unroll
    for (int mt = 0; mt < 4; ++mt) {
      int m = wm * 64 + mt * 16 + lr;
      const unsigned short* ph = &Ahi[cur][m * BK + half * 8];
      ah[mt].u[0] = *(const uint4*)(ph);
      ah[mt].u[1] = *(const uint4*)(ph + 16);
      const unsigned short* plo = &Alo[cur][m * BK + half * 8];
      al[mt].u[0] = *(const uint4*)(plo);
      al[mt].u[1] = *(const uint4*)(plo + 16);
    }

    // ---- B fragments: f32 from LDS, convert hi/lo here (co-exec w/ WMMA) --
    #pragma unroll
    for (int nt = 0; nt < 2; ++nt) {
      int n = wn * 32 + nt * 16 + lr;
      FragF bf;
      const float* pb = &Bf32[cur][n * BK + half * 8];
      bf.u[0] = ((const uint4*)pb)[0];
      bf.u[1] = ((const uint4*)pb)[1];
      bf.u[2] = ((const uint4*)pb)[4];
      bf.u[3] = ((const uint4*)pb)[5];
      FragB bh, bl;
      #pragma unroll
      for (int e = 0; e < 16; ++e) {
        unsigned short h = f2bf_rne(bf.f[e]);
        bh.s[e] = h;
        bl.s[e] = f2bf_rne(bf.f[e] - bf2f(h));
      }
      #pragma unroll
      for (int mt = 0; mt < 4; ++mt) {
        acc[mt][nt] = __builtin_amdgcn_wmma_f32_16x16x32_bf16(
            false, ah[mt].v, false, bh.v, (short)0, acc[mt][nt], false, false);
        acc[mt][nt] = __builtin_amdgcn_wmma_f32_16x16x32_bf16(
            false, ah[mt].v, false, bl.v, (short)0, acc[mt][nt], false, false);
        acc[mt][nt] = __builtin_amdgcn_wmma_f32_16x16x32_bf16(
            false, al[mt].v, false, bh.v, (short)0, acc[mt][nt], false, false);
      }
    }

    // ---- commit next B tile to LDS ----
    if (has_next) {
      #pragma unroll
      for (int i = 0; i < 4; ++i) {
        int k = b_q * 4 + i * 8;
        float4 v;
        v.x = breg[i * 4 + 0];
        v.y = breg[i * 4 + 1];
        v.z = breg[i * 4 + 2];
        v.w = breg[i * 4 + 3];
        *(float4*)(&Bf32[nxt][b_col * BK + k]) = v;
      }
    }

    // ---- close pipeline stage: A DMA done + all waves done with buffers ----
    if (wave == 0) __builtin_amdgcn_s_wait_tensorcnt(0);
    __syncthreads();
  }

  // ---- epilogue: D layout (lane<16: M=v, lane>=16: M=8+v; N=lane%16) ----
  float* ob = out + (size_t)b * COUT * TLEN;
  #pragma unroll
  for (int mt = 0; mt < 4; ++mt) {
    int orow0 = wm * 64 + mt * 16 + half * 8;      // relative to m0
    #pragma unroll
    for (int nt = 0; nt < 2; ++nt) {
      int t = t0 + wn * 32 + nt * 16 + lr;
      #pragma unroll
      for (int r = 0; r < 8; ++r) {
        int orel = orow0 + r;
        ob[(size_t)(m0 + orel) * TLEN + t] = acc[mt][nt][r] + biasS[orel];
      }
    }
  }
}

// ---------------------------------------------------------------------------
extern "C" void kernel_launch(void* const* d_in, const int* in_sizes, int n_in,
                              void* d_out, int out_size, void* d_ws, size_t ws_size,
                              hipStream_t stream) {
  const float* x  = (const float*)d_in[0];
  const float* dd = (const float*)d_in[1];
  const float* W0 = (const float*)d_in[2];
  const float* b0 = (const float*)d_in[3];
  const float* W1 = (const float*)d_in[4];
  const float* W2 = (const float*)d_in[5];
  float* out = (float*)d_out;

  // scratch: bf16 hi then bf16 lo for [W0;W1;W2]  (3 MB total)
  unsigned short* whi = (unsigned short*)d_ws;
  unsigned short* wlo = whi + (size_t)3 * COUT * CIN;

  const int prepN = 3 * COUT * CIN;
  prep_weights_kernel<<<(prepN + 255) / 256, 256, 0, stream>>>(W0, W1, W2, whi, wlo);

  dim3 grid(BATCH * (TLEN / BN), COUT / BM);
  adaptive_conv_wmma_kernel<<<grid, 256, 0, stream>>>(x, dd, whi, wlo, b0, out);
}
